// Mamba2Layer_47485158425275
// MI455X (gfx1250) — compile-verified, matrix operand order checked by reading, not run on hardware
//
#include <hip/hip_runtime.h>
#include <cstddef>
#include <cstdint>

// MI455X (gfx1250) Mamba2 forward. Memory-bound (~1 GB traffic @ 23.3 TB/s vs
// ~0.5 TFLOP): all global data stays fp32; matrix math on v_wmma_f32_16x16x32_f16
// with a hi/lo f16 split in the projection GEMMs (~fp32 accuracy, compute is free).
// Projection GEMMs use double-buffered global_load_async_to_lds_b128 DMA staging
// (ASYNCcnt-tracked) so global latency overlaps the WMMA pipeline.
// Workspace budget ~694 MB.

typedef __attribute__((ext_vector_type(16))) _Float16 v16h;
typedef __attribute__((ext_vector_type(8))) float v8f;

#define B_ 2
#define S_ 4096
#define DM 2048
#define DINNER 4096
#define DSSM 4096
#define NSTATE 128
#define NH 64
#define PD 64
#define NC 16
#define CH 256
#define CONVD 4352
#define DPROJ 8512
#define DTOFF 8448
#define MTOT 8192
#define EPS_ 1e-5f

__device__ __forceinline__ v8f vzero8() {
  v8f z;
#pragma unroll
  for (int i = 0; i < 8; ++i) z[i] = 0.0f;
  return z;
}

__device__ __forceinline__ v8f wmma_f16(v16h a, v16h b, v8f c) {
  // D = A(16x32 f16) * B(32x16 f16) + C(16x16 f32)
  return __builtin_amdgcn_wmma_f32_16x16x32_f16(false, a, false, b, (short)0, c,
                                                false, false);
}

// f16 operand fragment from an f16 LDS tile laid out [row][K] (leading dim ldk).
// Layout per ISA: lane r=lane&15 selects row; kh=(lane>>4)*8; halves 0..7 hold
// K=kh..kh+7 and halves 8..15 hold K=16+kh..16+kh+7.
__device__ __forceinline__ v16h ld_frag(const _Float16* base, int row, int ldk, int k0) {
  const int lane = threadIdx.x & 31;
  const int r = lane & 15;
  const int kh = (lane >> 4) << 3;
  const _Float16* p = base + (size_t)(row + r) * ldk + k0 + kh;
  v16h f;
#pragma unroll
  for (int j = 0; j < 8; ++j) f[j] = p[j];
#pragma unroll
  for (int j = 0; j < 8; ++j) f[8 + j] = p[16 + j];
  return f;
}

// hi/lo split fragment pair straight from a raw fp32 LDS tile [row][K].
__device__ __forceinline__ void ld_frag_split_rowK(const float* base, int row,
                                                   int ldk, v16h& hi, v16h& lo) {
  const int lane = threadIdx.x & 31;
  const float* p = base + (size_t)(row + (lane & 15)) * ldk + ((lane >> 4) << 3);
#pragma unroll
  for (int j = 0; j < 8; ++j) {
    float v0 = p[j];
    float v1 = p[16 + j];
    _Float16 h0 = (_Float16)v0;
    _Float16 h1 = (_Float16)v1;
    hi[j] = h0;
    lo[j] = (_Float16)(v0 - (float)h0);
    hi[8 + j] = h1;
    lo[8 + j] = (_Float16)(v1 - (float)h1);
  }
}

// hi/lo split B fragment pair from a raw fp32 LDS tile in natural [K][N] layout.
__device__ __forceinline__ void ld_frag_split_colN(const float* base, int ncol0,
                                                   int ldn, v16h& hi, v16h& lo) {
  const int lane = threadIdx.x & 31;
  const int kh = (lane >> 4) << 3;
  const float* p = base + (size_t)kh * ldn + ncol0 + (lane & 15);
#pragma unroll
  for (int j = 0; j < 8; ++j) {
    float v0 = p[(size_t)j * ldn];
    float v1 = p[(size_t)(16 + j) * ldn];
    _Float16 h0 = (_Float16)v0;
    _Float16 h1 = (_Float16)v1;
    hi[j] = h0;
    lo[j] = (_Float16)(v0 - (float)h0);
    hi[8 + j] = h1;
    lo[8 + j] = (_Float16)(v1 - (float)h1);
  }
}

// Async DMA: copy 16B from (sbase + voff) in global memory to LDS byte
// address lds_off. GVS addressing per CDNA5 ISA.
__device__ __forceinline__ void async_ld_b128(uint32_t lds_off, const float* sbase,
                                              uint32_t voff) {
  asm volatile("global_load_async_to_lds_b128 %0, %1, %2"
               :
               : "v"(lds_off), "v"(voff), "s"(sbase)
               : "memory");
}

// ---------------------------------------------------------------------------
// Split-f16 GEMM: C[M,N] = A[M,K] * B[K,N], all fp32 in memory.
// BM=128, BN=64, BK=32. 256 threads = 8 waves in a 4(m) x 2(n) wave grid,
// each wave owns a 32x32 tile = 2x2 WMMA tiles. Double-buffered async DMA
// stages raw fp32 tiles to LDS; fragments split to hi/lo f16 on load:
// acc = ah*bh + ah*bl + al*bh.  Requires M%128==0, N%64==0, K%32==0.
// ---------------------------------------------------------------------------
__global__ __launch_bounds__(256) void gemm_splitf16_kernel(
    const float* __restrict__ A, const float* __restrict__ Bm,
    float* __restrict__ C, int M, int N, int K) {
  // Staging strides keep every 16B DMA write 16B-aligned in LDS:
  // A: 36 floats/row (144B), B: 68 floats/row (272B).
  __shared__ __attribute__((aligned(16))) float stA[2][128][36];
  __shared__ __attribute__((aligned(16))) float stB[2][32][68];
  const int m0 = blockIdx.y * 128;
  const int n0 = blockIdx.x * 64;
  const int tid = threadIdx.x;
  const int lane = tid & 31;
  const int wid = tid >> 5;
  const int wm = (wid & 3) * 32;
  const int wn = (wid >> 2) * 32;

  const uint32_t ldsA = (uint32_t)(uintptr_t)&stA[0][0][0];
  const uint32_t ldsB = (uint32_t)(uintptr_t)&stB[0][0][0];

  // Per-thread DMA slots: A tile 128x32 -> 4 x b128 per thread,
  // B tile 32x64 -> 2 x b128 per thread. 6 async instructions per k-step.
  int ar[4], ac[4], br[2], bc[2];
#pragma unroll
  for (int i = 0; i < 4; ++i) {
    int q = tid + (i << 8);
    ar[i] = q >> 3;
    ac[i] = (q & 7) << 2;
  }
#pragma unroll
  for (int i = 0; i < 2; ++i) {
    int q = tid + (i << 8);
    br[i] = q >> 4;
    bc[i] = (q & 15) << 2;
  }

  auto issue = [&](int kk, int buf) {
#pragma unroll
    for (int i = 0; i < 4; ++i) {
      uint32_t l = ldsA + (uint32_t)((buf * 128 * 36 + ar[i] * 36 + ac[i]) << 2);
      uint32_t g = (uint32_t)((((size_t)(m0 + ar[i])) * K + kk + ac[i]) << 2);
      async_ld_b128(l, A, g);
    }
#pragma unroll
    for (int i = 0; i < 2; ++i) {
      uint32_t l = ldsB + (uint32_t)((buf * 32 * 68 + br[i] * 68 + bc[i]) << 2);
      uint32_t g = (uint32_t)((((size_t)(kk + br[i])) * N + n0 + bc[i]) << 2);
      async_ld_b128(l, Bm, g);
    }
  };

  v8f acc[2][2];
#pragma unroll
  for (int i = 0; i < 2; ++i)
#pragma unroll
    for (int j = 0; j < 2; ++j) acc[i][j] = vzero8();

  issue(0, 0);
  const int nk = K >> 5;
  for (int ik = 0; ik < nk; ++ik) {
    const int buf = ik & 1;
    if (ik + 1 < nk) {
      issue((ik + 1) << 5, buf ^ 1);  // prefetch next tile into other buffer
      asm volatile("s_wait_asynccnt 0x6" ::: "memory");  // current tile landed
    } else {
      asm volatile("s_wait_asynccnt 0x0" ::: "memory");
    }
    __syncthreads();  // all waves' DMA portions visible

    const float* pA = &stA[buf][0][0];
    const float* pB = &stB[buf][0][0];
    v16h ah[2], al[2], bh[2], bl[2];
#pragma unroll
    for (int t = 0; t < 2; ++t) {
      ld_frag_split_rowK(pA, wm + t * 16, 36, ah[t], al[t]);
      ld_frag_split_colN(pB, wn + t * 16, 68, bh[t], bl[t]);
    }
#pragma unroll
    for (int ti = 0; ti < 2; ++ti)
#pragma unroll
      for (int tj = 0; tj < 2; ++tj) {
        v8f t0 = wmma_f16(al[ti], bh[tj], acc[ti][tj]);
        v8f t1 = wmma_f16(ah[ti], bl[tj], t0);
        acc[ti][tj] = wmma_f16(ah[ti], bh[tj], t1);
      }
    __syncthreads();  // done reading buf before it is DMA-overwritten
  }

  const int ncol = lane & 15;
  const int moff = (lane >> 4) << 3;
#pragma unroll
  for (int ti = 0; ti < 2; ++ti)
#pragma unroll
    for (int tj = 0; tj < 2; ++tj) {
      float* cp =
          C + (size_t)(m0 + wm + ti * 16 + moff) * N + n0 + wn + tj * 16 + ncol;
#pragma unroll
      for (int j = 0; j < 8; ++j) cp[(size_t)j * N] = acc[ti][tj][j];
    }
}

// ---------------------------------------------------------------------------
// Depthwise causal conv (4 taps) + SiLU over the xBC slice of zxbcdt.
// ---------------------------------------------------------------------------
__global__ __launch_bounds__(256) void conv_silu_kernel(
    const float* __restrict__ zx, const float* __restrict__ cw,
    const float* __restrict__ cb, float* __restrict__ xBC) {
  const int NC4 = CONVD / 4;
  int idx = blockIdx.x * 256 + threadIdx.x;
  if (idx >= MTOT * NC4) return;
  int m = idx / NC4;
  int c4 = (idx % NC4) << 2;
  int b = m >> 12;
  int t = m & (S_ - 1);
  float4 bv = *(const float4*)(cb + c4);
  float acc[4] = {bv.x, bv.y, bv.z, bv.w};
#pragma unroll
  for (int i = 0; i < 4; ++i) {
    int tt = t + i - 3;
    if (tt < 0) continue;
    float4 xv = *(const float4*)(zx + (size_t)(b * S_ + tt) * DPROJ + DINNER + c4);
    float4 wv = *(const float4*)(cw + i * CONVD + c4);
    acc[0] += xv.x * wv.x;
    acc[1] += xv.y * wv.y;
    acc[2] += xv.z * wv.z;
    acc[3] += xv.w * wv.w;
  }
  float4 o;
  o.x = acc[0] / (1.f + expf(-acc[0]));
  o.y = acc[1] / (1.f + expf(-acc[1]));
  o.z = acc[2] / (1.f + expf(-acc[2]));
  o.w = acc[3] / (1.f + expf(-acc[3]));
  *(float4*)(xBC + (size_t)m * CONVD + c4) = o;
}

// ---------------------------------------------------------------------------
// dt = softplus(dt_raw + bias); dA = -exp(A_log)*dt; inclusive cumsum per chunk.
// One thread per (b, chunk, head) = 2048 threads total.
// ---------------------------------------------------------------------------
__global__ void dt_acum_kernel(const float* __restrict__ zx,
                               const float* __restrict__ dt_bias,
                               const float* __restrict__ A_log,
                               float* __restrict__ dtv, float* __restrict__ acum) {
  int idx = blockIdx.x * blockDim.x + threadIdx.x;
  if (idx >= B_ * NC * NH) return;
  int h = idx % NH;
  int c = (idx / NH) % NC;
  int b = idx / (NH * NC);
  float negA = -expf(A_log[h]);
  float bias = dt_bias[h];
  float s = 0.f;
  for (int l = 0; l < CH; ++l) {
    int row = b * S_ + c * CH + l;
    float x = zx[(size_t)row * DPROJ + DTOFF + h] + bias;
    float sp = log1pf(expf(-fabsf(x))) + fmaxf(x, 0.f);
    dtv[(size_t)row * NH + h] = sp;
    s += negA * sp;
    acum[(size_t)row * NH + h] = s;
  }
}

// ---------------------------------------------------------------------------
// SSD per (b, chunk, head): Y_diag = (L o (C*B^T)) * (x*dt), chunk states.
// ---------------------------------------------------------------------------
__global__ __launch_bounds__(256) void ssd_chunk_kernel(
    const float* __restrict__ xBC, const float* __restrict__ dtv,
    const float* __restrict__ acum, float* __restrict__ ybuf,
    float* __restrict__ states) {
  __shared__ _Float16 sB[CH][NSTATE + 8];   // B[l][n]
  __shared__ _Float16 sC[CH][NSTATE + 8];   // C[l][n]
  __shared__ _Float16 sXt[PD][CH + 8];      // (x*dt)^T : [p][l]
  __shared__ float sAc[CH];
  __shared__ float sDec[CH];
  __shared__ _Float16 sG[8][16][40];        // per-wave G' scratch
  const int blk = blockIdx.x;
  const int h = blk % NH;
  const int c = (blk / NH) % NC;
  const int b = blk / (NH * NC);
  const int tid = threadIdx.x;
  const int lane = tid & 31;
  const int wid = tid >> 5;
  const int row0 = b * S_ + c * CH;

  for (int q = tid; q < CH * (NSTATE / 4); q += 256) {
    int l = q >> 5;
    int n4 = (q & 31) << 2;
    const float* src = xBC + (size_t)(row0 + l) * CONVD + DSSM + n4;
    float4 v = *(const float4*)src;
    sB[l][n4 + 0] = (_Float16)v.x;
    sB[l][n4 + 1] = (_Float16)v.y;
    sB[l][n4 + 2] = (_Float16)v.z;
    sB[l][n4 + 3] = (_Float16)v.w;
    float4 w = *(const float4*)(src + NSTATE);
    sC[l][n4 + 0] = (_Float16)w.x;
    sC[l][n4 + 1] = (_Float16)w.y;
    sC[l][n4 + 2] = (_Float16)w.z;
    sC[l][n4 + 3] = (_Float16)w.w;
  }
  for (int q = tid; q < CH * (PD / 4); q += 256) {
    int l = q >> 4;
    int p4 = (q & 15) << 2;
    float dt = dtv[(size_t)(row0 + l) * NH + h];
    float4 v = *(const float4*)(xBC + (size_t)(row0 + l) * CONVD + h * PD + p4);
    sXt[p4 + 0][l] = (_Float16)(v.x * dt);
    sXt[p4 + 1][l] = (_Float16)(v.y * dt);
    sXt[p4 + 2][l] = (_Float16)(v.z * dt);
    sXt[p4 + 3][l] = (_Float16)(v.w * dt);
  }
  for (int l = tid; l < CH; l += 256) sAc[l] = acum[(size_t)(row0 + l) * NH + h];
  __syncthreads();
  const float alast = sAc[CH - 1];
  for (int l = tid; l < CH; l += 256) sDec[l] = expf(alast - sAc[l]);
  __syncthreads();

  const int ncol = lane & 15;
  const int moff = (lane >> 4) << 3;

  // Phase 1: Y_diag. Wave handles l-tiles {wid, wid+8}.
#pragma unroll
  for (int rep = 0; rep < 2; ++rep) {
    const int i = wid + rep * 8;
    v8f Y[4];
#pragma unroll
    for (int t = 0; t < 4; ++t) Y[t] = vzero8();
    for (int j2 = 0; j2 <= (i >> 1); ++j2) {
#pragma unroll
      for (int half = 0; half < 2; ++half) {
        const int j = 2 * j2 + half;
        if (j <= i) {
          v8f g = vzero8();
#pragma unroll
          for (int ks = 0; ks < 4; ++ks) {
            v16h a = ld_frag(&sC[0][0], i * 16, NSTATE + 8, ks * 32);
            v16h bb = ld_frag(&sB[0][0], j * 16, NSTATE + 8, ks * 32);
            g = wmma_f16(a, bb, g);
          }
          const int scol = j * 16 + ncol;
          const float as = sAc[scol];
#pragma unroll
          for (int e = 0; e < 8; ++e) {
            const int lrow = i * 16 + moff + e;
            float val = (lrow >= scol) ? g[e] * expf(sAc[lrow] - as) : 0.f;
            sG[wid][moff + e][half * 16 + ncol] = (_Float16)val;
          }
        } else {
#pragma unroll
          for (int e = 0; e < 8; ++e)
            sG[wid][moff + e][half * 16 + ncol] = (_Float16)0.f;
        }
      }
      // Same-wave LDS RAW (in-order DS pipe); belt-and-braces wait.
      asm volatile("s_wait_dscnt 0" ::: "memory");
      v16h ag = ld_frag(&sG[wid][0][0], 0, 40, 0);
#pragma unroll
      for (int t = 0; t < 4; ++t) {
        v16h bx = ld_frag(&sXt[0][0], t * 16, CH + 8, j2 * 32);
        Y[t] = wmma_f16(ag, bx, Y[t]);
      }
    }
#pragma unroll
    for (int t = 0; t < 4; ++t) {
      float* yp = ybuf + (size_t)(row0 + i * 16 + moff) * DSSM + h * PD + t * 16 + ncol;
#pragma unroll
      for (int e = 0; e < 8; ++e) yp[(size_t)e * DSSM] = Y[t][e];
    }
  }

  // Phase 2: states[n][p] = sum_l B[l][n]*dec[l]*(x*dt)[l][p]. Wave owns 16 n-rows.
  {
    const int nbase = wid * 16;
    v8f Sacc[4];
#pragma unroll
    for (int t = 0; t < 4; ++t) Sacc[t] = vzero8();
    const int n = nbase + ncol;
    const int kh = (lane >> 4) << 3;
#pragma unroll
    for (int ks = 0; ks < 8; ++ks) {
      v16h a;
#pragma unroll
      for (int j = 0; j < 8; ++j) {
        int l0 = ks * 32 + kh + j;
        int l1 = l0 + 16;
        a[j] = (_Float16)((float)sB[l0][n] * sDec[l0]);
        a[8 + j] = (_Float16)((float)sB[l1][n] * sDec[l1]);
      }
#pragma unroll
      for (int t = 0; t < 4; ++t) {
        v16h bx = ld_frag(&sXt[0][0], t * 16, CH + 8, ks * 32);
        Sacc[t] = wmma_f16(a, bx, Sacc[t]);
      }
    }
    const size_t base = ((size_t)(b * NC + c) * NH + h) * NSTATE * PD;
#pragma unroll
    for (int t = 0; t < 4; ++t)
#pragma unroll
      for (int e = 0; e < 8; ++e)
        states[base + (size_t)(nbase + moff + e) * PD + t * 16 + ncol] = Sacc[t][e];
  }
}

// ---------------------------------------------------------------------------
// Inter-chunk scan: P[c] = P[c-1]*exp(Atot[c-1]) + states_ch[c-1], P[0]=0.
// ---------------------------------------------------------------------------
__global__ void chunk_scan_kernel(const float* __restrict__ states_ch,
                                  const float* __restrict__ acum,
                                  float* __restrict__ states_pref) {
  int idx = blockIdx.x * 256 + threadIdx.x;
  if (idx >= B_ * NH * NSTATE * PD) return;
  int p = idx & (PD - 1);
  int n = (idx >> 6) & (NSTATE - 1);
  int h = (idx >> 13) & (NH - 1);
  int b = idx >> 19;
  float pref = 0.f;
  for (int c = 0; c < NC; ++c) {
    size_t off = ((size_t)(b * NC + c) * NH + h) * NSTATE * PD + (size_t)n * PD + p;
    states_pref[off] = pref;
    float atot = acum[(size_t)(b * S_ + c * CH + CH - 1) * NH + h];
    pref = pref * expf(atot) + states_ch[off];
  }
}

// ---------------------------------------------------------------------------
// Y += exp(acum[l]) * (C @ P) + x*D per (b, chunk, head).
// ---------------------------------------------------------------------------
__global__ __launch_bounds__(256) void yoff_kernel(
    const float* __restrict__ xBC, const float* __restrict__ acum,
    const float* __restrict__ states_pref, const float* __restrict__ Dv,
    float* __restrict__ ybuf) {
  __shared__ _Float16 sC[CH][NSTATE + 8];
  __shared__ _Float16 sPt[PD][NSTATE + 8];  // P^T : [p][n]
  __shared__ float sAc[CH];
  const int blk = blockIdx.x;
  const int h = blk % NH;
  const int c = (blk / NH) % NC;
  const int b = blk / (NH * NC);
  const int tid = threadIdx.x;
  const int lane = tid & 31;
  const int wid = tid >> 5;
  const int row0 = b * S_ + c * CH;
  for (int q = tid; q < CH * (NSTATE / 4); q += 256) {
    int l = q >> 5;
    int n4 = (q & 31) << 2;
    float4 v = *(const float4*)(xBC + (size_t)(row0 + l) * CONVD + DSSM + NSTATE + n4);
    sC[l][n4 + 0] = (_Float16)v.x;
    sC[l][n4 + 1] = (_Float16)v.y;
    sC[l][n4 + 2] = (_Float16)v.z;
    sC[l][n4 + 3] = (_Float16)v.w;
  }
  const size_t pbase = ((size_t)(b * NC + c) * NH + h) * NSTATE * PD;
  for (int q = tid; q < NSTATE * (PD / 4); q += 256) {
    int n = q >> 4;
    int p4 = (q & 15) << 2;
    float4 v = *(const float4*)(states_pref + pbase + (size_t)n * PD + p4);
    sPt[p4 + 0][n] = (_Float16)v.x;
    sPt[p4 + 1][n] = (_Float16)v.y;
    sPt[p4 + 2][n] = (_Float16)v.z;
    sPt[p4 + 3][n] = (_Float16)v.w;
  }
  for (int l = tid; l < CH; l += 256) sAc[l] = acum[(size_t)(row0 + l) * NH + h];
  __syncthreads();
  const float Dh = Dv[h];
  const int ncol = lane & 15;
  const int moff = (lane >> 4) << 3;
#pragma unroll
  for (int rep = 0; rep < 2; ++rep) {
    const int i = wid + rep * 8;
    v8f Y[4];
#pragma unroll
    for (int t = 0; t < 4; ++t) Y[t] = vzero8();
#pragma unroll
    for (int ks = 0; ks < 4; ++ks) {
      v16h a = ld_frag(&sC[0][0], i * 16, NSTATE + 8, ks * 32);
#pragma unroll
      for (int t = 0; t < 4; ++t) {
        v16h bp = ld_frag(&sPt[0][0], t * 16, NSTATE + 8, ks * 32);
        Y[t] = wmma_f16(a, bp, Y[t]);
      }
    }
#pragma unroll
    for (int t = 0; t < 4; ++t)
#pragma unroll
      for (int e = 0; e < 8; ++e) {
        int l = i * 16 + moff + e;
        int p = t * 16 + ncol;
        size_t yo = (size_t)(row0 + l) * DSSM + h * PD + p;
        float x = xBC[(size_t)(row0 + l) * CONVD + h * PD + p];
        ybuf[yo] = ybuf[yo] + expf(sAc[l]) * Y[t][e] + x * Dh;
      }
  }
}

// ---------------------------------------------------------------------------
// y = y*silu(z); RMS-norm over 4096 dims; * norm_scale. One block per row.
// ---------------------------------------------------------------------------
__global__ __launch_bounds__(256) void gate_norm_kernel(
    const float* __restrict__ zx, const float* __restrict__ nscale,
    float* __restrict__ ybuf) {
  __shared__ float red[256];
  const int m = blockIdx.x;
  const int tid = threadIdx.x;
  const float* zrow = zx + (size_t)m * DPROJ;
  float* yrow = ybuf + (size_t)m * DSSM;
  float g[16];
  float ss = 0.f;
#pragma unroll
  for (int i = 0; i < 16; ++i) {
    int d = tid + (i << 8);
    float z = zrow[d];
    float y = yrow[d] * (z / (1.f + expf(-z)));
    g[i] = y;
    ss += y * y;
  }
  red[tid] = ss;
  __syncthreads();
  for (int s = 128; s > 0; s >>= 1) {
    if (tid < s) red[tid] += red[tid + s];
    __syncthreads();
  }
  const float scale = rsqrtf(red[0] / (float)DSSM + EPS_);
#pragma unroll
  for (int i = 0; i < 16; ++i) {
    int d = tid + (i << 8);
    yrow[d] = g[i] * scale * nscale[d];
  }
}

extern "C" void kernel_launch(void* const* d_in, const int* in_sizes, int n_in,
                              void* d_out, int out_size, void* d_ws, size_t ws_size,
                              hipStream_t stream) {
  (void)in_sizes; (void)n_in; (void)out_size; (void)ws_size;
  const float* u = (const float*)d_in[0];
  const float* W_in = (const float*)d_in[1];
  const float* conv_w = (const float*)d_in[2];
  const float* conv_b = (const float*)d_in[3];
  const float* dt_bias = (const float*)d_in[4];
  const float* A_log = (const float*)d_in[5];
  const float* Dv = (const float*)d_in[6];
  const float* norm_scale = (const float*)d_in[7];
  const float* W_out = (const float*)d_in[8];
  float* out = (float*)d_out;

  float* zx = (float*)d_ws;                                   // [8192][8512]
  float* xBC = zx + (size_t)MTOT * DPROJ;                     // [8192][4352]
  float* dtv = xBC + (size_t)MTOT * CONVD;                    // [8192][64]
  float* acum = dtv + (size_t)MTOT * NH;                      // [8192][64]
  float* st_ch = acum + (size_t)MTOT * NH;                    // [2][16][64][128][64]
  float* st_pref = st_ch + (size_t)B_ * NC * NH * NSTATE * PD;
  float* ybuf = st_pref + (size_t)B_ * NC * NH * NSTATE * PD; // [8192][4096]

  gemm_splitf16_kernel<<<dim3(DPROJ / 64, MTOT / 128), 256, 0, stream>>>(
      u, W_in, zx, MTOT, DPROJ, DM);
  conv_silu_kernel<<<(MTOT * (CONVD / 4) + 255) / 256, 256, 0, stream>>>(
      zx, conv_w, conv_b, xBC);
  dt_acum_kernel<<<(B_ * NC * NH + 255) / 256, 256, 0, stream>>>(
      zx, dt_bias, A_log, dtv, acum);
  ssd_chunk_kernel<<<B_ * NC * NH, 256, 0, stream>>>(xBC, dtv, acum, ybuf, st_ch);
  chunk_scan_kernel<<<(B_ * NH * NSTATE * PD) / 256, 256, 0, stream>>>(
      st_ch, acum, st_pref);
  yoff_kernel<<<B_ * NC * NH, 256, 0, stream>>>(xBC, acum, st_pref, Dv, ybuf);
  gate_norm_kernel<<<MTOT, 256, 0, stream>>>(zx, norm_scale, ybuf);
  gemm_splitf16_kernel<<<dim3(DM / 64, MTOT / 128), 256, 0, stream>>>(
      ybuf, W_out, out, MTOT, DM, DSSM);
}
